// Triaffine_24850680775240
// MI455X (gfx1250) — matrix operand order repaired and checked by reading, not compiled, not run on hardware
//
#include <hip/hip_runtime.h>
#include <hip/hip_bf16.h>

// ---------------------------------------------------------------------------
// Triaffine  out[b,z,x,y] = sum_{i,j,k} xb[b,x,i] z[b,z,k] W[i,k,j] yb[b,y,j]
// B=8, S=128, D=512.  3-stage bf16 WMMA pipeline (v_wmma_f32_16x16x32_bf16):
//   P0: repack inputs to bf16 (xb/yb with bias=1 at col 512, pad->544)
//   P1: LDS-tiled transpose W[i,k,j] f32 -> Wt[j][i][k] bf16 (i padded 544)
//   K12 (one WG per (b,j)): M_j[t][i] = Wt[j] x Zb[b]^T into LDS (138KB),
//       then Q[b][t][x][j] = Xb[b] x M_j   (stage-1 GEMM is 75% of all FLOPs)
//   K3 (one WG per (b,t)): out[b][t] = Q[b][t] x Yb[b]^T
// ws layout: Wt (286MB) | Q (143MB) | Xb | Yb | Zb   (~432 MB total)
// ---------------------------------------------------------------------------

typedef __attribute__((ext_vector_type(16))) __bf16 v16bf;
typedef __attribute__((ext_vector_type(8)))  __bf16 v8bf;
typedef __attribute__((ext_vector_type(8)))  float  v8f;

#define NB   8
#define NS   128
#define ND   512
#define NDP1 513
#define KPAD 544   // i (and j) dim padded to multiple of 32 for K-steps
#define MLP  552   // LDS pitch for M_j rows (bf16 elems, 16B-aligned rows)

#define WT_BYTES ((unsigned long long)NDP1 * KPAD * ND * 2ull)          // 285.9 MB
#define Q_BYTES  ((unsigned long long)NB * NS * NS * KPAD * 2ull)       // 142.6 MB
#define XB_BYTES ((unsigned long long)NB * NS * KPAD * 2ull)
#define Q_OFF    (WT_BYTES)
#define XB_OFF   (Q_OFF + Q_BYTES)
#define YB_OFF   (XB_OFF + XB_BYTES)
#define ZB_OFF   (YB_OFF + XB_BYTES)

__device__ __forceinline__ unsigned short f2bf(float f) {
  unsigned u = __float_as_uint(f);
  u += 0x7FFFu + ((u >> 16) & 1u);        // round-to-nearest-even
  return (unsigned short)(u >> 16);
}
__device__ __forceinline__ __bf16 tobf(float f) {
  unsigned short u = f2bf(f);
  return __builtin_bit_cast(__bf16, u);
}
__device__ __forceinline__ unsigned pk2(float lo, float hi) {
  return (unsigned)f2bf(lo) | ((unsigned)f2bf(hi) << 16);
}

// A operand, 16x32 bf16, row-major source (pitch in elems, contiguous K):
// lanes 0-15: M=lane, K runs {k0..k0+7, k0+16..k0+23};
// lanes 16-31: M=lane-16, K runs {k0+8..+15, k0+24..+31}. Two 16B loads.
__device__ __forceinline__ v16bf loadA(const __bf16* base, int pitch, int lane) {
  int m  = lane & 15;
  int ko = (lane < 16) ? 0 : 8;
  const __bf16* p = base + (long long)m * pitch + ko;
  v8bf lo = *(const v8bf*)p;
  v8bf hi = *(const v8bf*)(p + 16);
  v16bf r;
#pragma unroll
  for (int t = 0; t < 8; ++t) { r[t] = lo[t]; r[8 + t] = hi[t]; }
  return r;
}

// B operand, 32x16 bf16; column N contiguous in K (i.e. source row = column):
// lanes 0-15: N=lane, K runs {k0..k0+7, k0+16..k0+23};
// lanes 16-31: N=lane-16, K runs {k0+8..+15, k0+24..+31}.
__device__ __forceinline__ v16bf loadB(const __bf16* base, int pitch, int lane) {
  int n  = lane & 15;
  int ko = (lane < 16) ? 0 : 8;
  const __bf16* p = base + (long long)n * pitch + ko;
  v8bf lo = *(const v8bf*)p;
  v8bf hi = *(const v8bf*)(p + 16);
  v16bf r;
#pragma unroll
  for (int t = 0; t < 8; ++t) { r[t] = lo[t]; r[8 + t] = hi[t]; }
  return r;
}

__device__ __forceinline__ v8f wmma_bf16(v16bf a, v16bf b, v8f c) {
  return __builtin_amdgcn_wmma_f32_16x16x32_bf16(false, a, false, b,
                                                 (short)0, c, false, false);
}

// ---------------- P0: bf16 repack of x, y (bias+pad) and z -----------------
__global__ void prep_inputs(const float* __restrict__ x, const float* __restrict__ y,
                            const float* __restrict__ z,
                            __bf16* __restrict__ Xb, __bf16* __restrict__ Yb,
                            __bf16* __restrict__ Zb) {
  long long stride = (long long)gridDim.x * blockDim.x;
  long long i0 = (long long)blockIdx.x * blockDim.x + threadIdx.x;
  long long nPad = (long long)NB * NS * KPAD;
  for (long long t = i0; t < nPad; t += stride) {
    long long row = t / KPAD;
    int c = (int)(t % KPAD);
    float xv = (c < ND) ? x[row * ND + c] : (c == ND ? 1.0f : 0.0f);
    float yv = (c < ND) ? y[row * ND + c] : (c == ND ? 1.0f : 0.0f);
    Xb[t] = tobf(xv);
    Yb[t] = tobf(yv);
  }
  long long nZ = (long long)NB * NS * ND;
  for (long long t = i0; t < nZ; t += stride) Zb[t] = tobf(z[t]);
}

// ---------------- P1: W[i,k,j] f32 -> Wt[j][i][k] bf16 (tiled transpose) ---
__global__ void transpose_w(const float* __restrict__ W, __bf16* __restrict__ Wt) {
  __shared__ float tile[32][33];
  int i  = blockIdx.z;            // 0..512
  int j0 = blockIdx.x * 32;       // 17 tiles (covers 513)
  int k0 = blockIdx.y * 32;       // 16 tiles (512 exact)
  int cc = threadIdx.x & 31;
  int r0 = threadIdx.x >> 5;      // 0..7
#pragma unroll
  for (int rr = 0; rr < 32; rr += 8) {
    int k = k0 + r0 + rr;
    int j = j0 + cc;              // j contiguous per 32-lane row: coalesced
    float v = (j < NDP1) ? W[((long long)i * ND + k) * NDP1 + j] : 0.0f;
    tile[r0 + rr][cc] = v;
  }
  __syncthreads();
#pragma unroll
  for (int rr = 0; rr < 32; rr += 8) {
    int j = j0 + r0 + rr;
    int k = k0 + cc;              // k contiguous per row: coalesced bf16 write
    if (j < NDP1)
      Wt[((long long)j * KPAD + i) * ND + k] = tobf(tile[cc][r0 + rr]);
  }
}

// zero the padded rows i in [513,544) of Wt so stage-1 pad rows become 0
__global__ void zero_wt_pad(__bf16* __restrict__ Wt) {
  long long stride = (long long)gridDim.x * blockDim.x;
  long long total = (long long)NDP1 * (KPAD - NDP1) * ND;
  for (long long t = (long long)blockIdx.x * blockDim.x + threadIdx.x;
       t < total; t += stride) {
    long long j = t / ((KPAD - NDP1) * ND);
    long long r = t % ((KPAD - NDP1) * ND);
    long long i = NDP1 + r / ND;
    long long k = r % ND;
    Wt[(j * KPAD + i) * ND + k] = __builtin_bit_cast(__bf16, (unsigned short)0);
  }
}

// ---------------- K12: fused stage 1+2, one WG per (b, j) ------------------
__global__ void __launch_bounds__(256)
triaffine_stage12(const __bf16* __restrict__ Wt, const __bf16* __restrict__ Zb,
                  const __bf16* __restrict__ Xb, __bf16* __restrict__ Q) {
  __shared__ __align__(16) __bf16 Mlds[NS * MLP];   // 138 KB of 320 KB LDS
  const int j    = blockIdx.x;
  const int b    = blockIdx.y;
  const int lane = threadIdx.x & 31;
  const int wave = threadIdx.x >> 5;                // 8 waves

  const __bf16* Wj  = Wt + (long long)j * KPAD * ND;   // [i][k], pitch ND
  const __bf16* Zbb = Zb + (long long)b * NS * ND;     // [t][k], pitch ND

  // Phase 1: Mlds[t][i] = sum_k Wt[j][i][k] * Zb[b][t][k]   (M=i, N=t, K=k)
  for (int it = wave; it < KPAD / 16; it += 8) {     // 34 i-tiles striped
    v8f acc[8] = {};                                 // 8 t-tiles resident
    for (int k0 = 0; k0 < ND; k0 += 32) {
      v16bf a = loadA(Wj + it * 16 * ND + k0, ND, lane);   // reused 8x
#pragma unroll
      for (int n = 0; n < 8; ++n) {
        v16bf bb = loadB(Zbb + n * 16 * ND + k0, ND, lane);
        acc[n] = wmma_bf16(a, bb, acc[n]);
      }
    }
    int m0 = (lane < 16) ? 0 : 8;                    // C rows held by lane
#pragma unroll
    for (int n = 0; n < 8; ++n) {
      int t = n * 16 + (lane & 15);
      uint4 p;
      p.x = pk2(acc[n][0], acc[n][1]);
      p.y = pk2(acc[n][2], acc[n][3]);
      p.z = pk2(acc[n][4], acc[n][5]);
      p.w = pk2(acc[n][6], acc[n][7]);
      *(uint4*)&Mlds[t * MLP + it * 16 + m0] = p;    // 16B aligned ds_store
    }
  }
  __syncthreads();

  // Phase 2: Q[b][t][x][j] = sum_i Xb[b][x][i] * Mlds[t][i]  (M=x, N=t, K=i)
  const __bf16* Xbb = Xb + (long long)b * NS * KPAD;
  v8f acc[8] = {};
  for (int k0 = 0; k0 < KPAD; k0 += 32) {
    v16bf a = loadA(Xbb + wave * 16 * KPAD + k0, KPAD, lane);
#pragma unroll
    for (int n = 0; n < 8; ++n) {
      v16bf bb = loadB(&Mlds[n * 16 * MLP + k0], MLP, lane);
      acc[n] = wmma_bf16(a, bb, acc[n]);
    }
  }
  int m0 = (lane < 16) ? 0 : 8;
#pragma unroll
  for (int n = 0; n < 8; ++n) {
    int t = n * 16 + (lane & 15);
#pragma unroll
    for (int r = 0; r < 8; ++r) {
      int xx = wave * 16 + m0 + r;
      Q[(((long long)b * NS + t) * NS + xx) * KPAD + j] = tobf(acc[n][r]);
    }
  }
}

// ---------------- K3: one WG per (b, t): out[b][t] = Q[b][t] x Yb^T --------
__global__ void __launch_bounds__(256)
triaffine_stage3(const __bf16* __restrict__ Q, const __bf16* __restrict__ Yb,
                 float* __restrict__ out) {
  const int t    = blockIdx.x;
  const int b    = blockIdx.y;
  const int lane = threadIdx.x & 31;
  const int wave = threadIdx.x >> 5;
  const __bf16* Arow = Q + ((long long)b * NS + t) * NS * KPAD;  // [x][j]
  const __bf16* Ybb  = Yb + (long long)b * NS * KPAD;            // [y][j]

  v8f acc[8] = {};
  for (int k0 = 0; k0 < KPAD; k0 += 32) {   // Yb pad cols are 0 -> j pad safe
    v16bf a = loadA(Arow + wave * 16 * KPAD + k0, KPAD, lane);
#pragma unroll
    for (int n = 0; n < 8; ++n) {
      v16bf bb = loadB(Ybb + n * 16 * KPAD + k0, KPAD, lane);
      acc[n] = wmma_bf16(a, bb, acc[n]);
    }
  }
  int m0 = (lane < 16) ? 0 : 8;
#pragma unroll
  for (int n = 0; n < 8; ++n) {
    int y = n * 16 + (lane & 15);
#pragma unroll
    for (int r = 0; r < 8; ++r) {
      int xx = wave * 16 + m0 + r;
      out[(((long long)b * NS + t) * NS + xx) * NS + y] = acc[n][r];
    }
  }
}

extern "C" void kernel_launch(void* const* d_in, const int* in_sizes, int n_in,
                              void* d_out, int out_size, void* d_ws, size_t ws_size,
                              hipStream_t stream) {
  (void)in_sizes; (void)n_in; (void)out_size; (void)ws_size;
  const float* x = (const float*)d_in[0];
  const float* y = (const float*)d_in[1];
  const float* z = (const float*)d_in[2];
  const float* w = (const float*)d_in[3];
  char* ws = (char*)d_ws;
  __bf16* Wt = (__bf16*)(ws);
  __bf16* Qw = (__bf16*)(ws + Q_OFF);
  __bf16* Xb = (__bf16*)(ws + XB_OFF);
  __bf16* Yb = (__bf16*)(ws + YB_OFF);
  __bf16* Zb = (__bf16*)(ws + ZB_OFF);

  prep_inputs<<<1024, 256, 0, stream>>>(x, y, z, Xb, Yb, Zb);
  transpose_w<<<dim3(17, 16, NDP1), 256, 0, stream>>>(w, Wt);
  zero_wt_pad<<<512, 256, 0, stream>>>(Wt);
  // j-major grid: co-resident WGs share the same Wt slice across batches (L2)
  triaffine_stage12<<<dim3(NDP1, NB), 256, 0, stream>>>(Wt, Zb, Xb, Qw);
  triaffine_stage3<<<dim3(NS, NB), 256, 0, stream>>>(Qw, Yb, (float*)d_out);
}
